// MultiheadAttention_46746424050430
// MI455X (gfx1250) — matrix operand order repaired and checked by reading, not compiled, not run on hardware
//
#include <hip/hip_runtime.h>

typedef __attribute__((ext_vector_type(16))) _Float16 v16h;
typedef __attribute__((ext_vector_type(8)))  _Float16 v8h;
typedef __attribute__((ext_vector_type(4)))  _Float16 v4h;
typedef __attribute__((ext_vector_type(8)))  float    v8f;
typedef __attribute__((ext_vector_type(4)))  float    v4f;
typedef __attribute__((ext_vector_type(4)))  unsigned int u32x4;
typedef __attribute__((ext_vector_type(8)))  int      i32x8;
typedef __attribute__((ext_vector_type(4)))  int      i32x4;

#define WMMA_F16(a, b, c) \
  __builtin_amdgcn_wmma_f32_16x16x32_f16(false, (a), false, (b), (short)0, (c), false, false)

#if defined(__HIP_DEVICE_COMPILE__) && __has_builtin(__builtin_amdgcn_tensor_load_to_lds)
#define USE_TDM 1
#else
#define USE_TDM 0
#endif

static constexpr int Bsz = 2, Ssz = 2048, Tsz = 2048, CH = 1024, NH = 16, DH = 64;
static constexpr int ROWS = Bsz * Ssz;  // 4096

#if USE_TDM
// 2D TDM tile load: tile_d0 x tile_d1 f16, global row stride CH elements,
// LDS rows padded from tile_d0*2 bytes by pad_amount dwords every
// 2^(pad_interval_code+1) dwords. D# per CDNA5 ISA.
__device__ inline void tdm_load_2d_f16(const _Float16* gptr, unsigned lds_byte,
                                       int tile_d0, int tile_d1,
                                       int pad_interval_code, int pad_amount_code) {
  const unsigned long long ga = (unsigned long long)(const void*)gptr;
  u32x4 g0;
  g0[0] = 1u;                                   // count=1, is_restore=0, gather=0
  g0[1] = lds_byte;                             // lds_addr (bytes)
  g0[2] = (unsigned)ga;                         // global_addr[31:0]
  g0[3] = (unsigned)((ga >> 32) & 0x01FFFFFFu)  // global_addr[56:32]
          | (2u << 30);                         // type=2 ("image")
  i32x8 g1;
  g1[0] = (1 << 16)                             // data_size = 2 bytes
          | (1 << 20)                           // pad_enable
          | (pad_interval_code << 22)           // dwords before pad = 2^(c+1)
          | (pad_amount_code << 25);            // pad dwords = c+1
  g1[1] = (tile_d0 << 16);                      // tensor_dim0 low16 @ bits 63:48
  g1[2] = (tile_d1 << 16);                      // tensor_dim1 low16 @ bits 95:80
  g1[3] = (tile_d0 << 16);                      // tile_dim0   @ bits 127:112
  g1[4] = tile_d1;                              // tile_dim1; tile_dim2=0
  g1[5] = CH;                                   // tensor_dim0_stride (elements)
  g1[6] = 0;
  g1[7] = 0;
  const i32x4 z4 = {0, 0, 0, 0};
  const i32x8 z8 = {0, 0, 0, 0, 0, 0, 0, 0};
  __builtin_amdgcn_tensor_load_to_lds(g0, g1, z4, z4, z8, 0);
}
#endif

// ---------------- WMMA fragment loaders (CDNA5 wave32 layouts) ----------------
__device__ inline v16h frag_a(const _Float16* p0, int ld, int lane) {
  const _Float16* p = p0 + (unsigned)(lane & 15) * (unsigned)ld + ((lane & 16) ? 16 : 0);
  v16h a;
#pragma unroll
  for (int j = 0; j < 16; ++j) a[j] = p[j];
  return a;
}
__device__ inline v16h frag_b(const _Float16* p0, int ld, int lane) {
  const _Float16* p = p0 + (unsigned)(lane & 15) * (unsigned)ld + ((lane & 16) ? 16 : 0);
  v16h b;
#pragma unroll
  for (int j = 0; j < 16; ++j) b[j] = p[j];
  return b;
}

// ---------------- Projection GEMM: f32 X @ f32 W + b -> f16 ----------------
__global__ __launch_bounds__(256) void k_gemm_qkv(const float* __restrict__ X,
                                                  const float* __restrict__ W,
                                                  const float* __restrict__ bias,
                                                  _Float16* __restrict__ Y) {
  constexpr int BK = 32, ALD = 40, BLD = 40;
  __shared__ _Float16 sA[128 * ALD];  // [m][k]
  __shared__ _Float16 sB[128 * BLD];  // [n][k]
  const int tid = threadIdx.x, lane = tid & 31, wid = tid >> 5;
  const int wr = wid >> 1, wc = wid & 1;
  const int m0 = blockIdx.y * 128, n0 = blockIdx.x * 128;

  v8f acc[2][4];
#pragma unroll
  for (int i = 0; i < 2; ++i)
#pragma unroll
    for (int j = 0; j < 4; ++j)
#pragma unroll
      for (int r = 0; r < 8; ++r) acc[i][j][r] = 0.f;

  const int ar = tid >> 3, ac = (tid & 7) * 4;
  const int bkr = tid >> 5, bnc = (tid & 31) * 4;

  for (int k0 = 0; k0 < CH; k0 += BK) {
    __syncthreads();
#pragma unroll
    for (int rr = 0; rr < 4; ++rr) {
      const int r = ar + rr * 32;
      const v4f x = *(const v4f*)(X + (unsigned)((m0 + r) * CH + k0 + ac));
      v4h h;
      h[0] = (_Float16)x[0]; h[1] = (_Float16)x[1];
      h[2] = (_Float16)x[2]; h[3] = (_Float16)x[3];
      *(v4h*)(sA + r * ALD + ac) = h;
    }
#pragma unroll
    for (int pp = 0; pp < 4; ++pp) {
      const int kk = bkr + pp * 8;
      const v4f w = *(const v4f*)(W + (unsigned)((k0 + kk) * CH + n0 + bnc));
      sB[(bnc + 0) * BLD + kk] = (_Float16)w[0];
      sB[(bnc + 1) * BLD + kk] = (_Float16)w[1];
      sB[(bnc + 2) * BLD + kk] = (_Float16)w[2];
      sB[(bnc + 3) * BLD + kk] = (_Float16)w[3];
    }
    __syncthreads();

    v16h aF[2], bF[4];
#pragma unroll
    for (int mt = 0; mt < 2; ++mt) aF[mt] = frag_a(sA + (wr * 32 + mt * 16) * ALD, ALD, lane);
#pragma unroll
    for (int nt = 0; nt < 4; ++nt) bF[nt] = frag_b(sB + (wc * 64 + nt * 16) * BLD, BLD, lane);
#pragma unroll
    for (int mt = 0; mt < 2; ++mt)
#pragma unroll
      for (int nt = 0; nt < 4; ++nt) acc[mt][nt] = WMMA_F16(aF[mt], bF[nt], acc[mt][nt]);
  }

  const int hf = (lane & 16) ? 8 : 0, nl = lane & 15;
#pragma unroll
  for (int mt = 0; mt < 2; ++mt)
#pragma unroll
    for (int nt = 0; nt < 4; ++nt) {
      const int col = n0 + wc * 64 + nt * 16 + nl;
      const float bs = bias[col];
#pragma unroll
      for (int r = 0; r < 8; ++r) {
        const int row = m0 + wr * 32 + mt * 16 + r + hf;
        Y[(unsigned)(row * CH + col)] = (_Float16)(acc[mt][nt][r] + bs);
      }
    }
}

// ---------------- Flash attention over f16 Q/K/V (software-pipelined) ----------------
__global__ __launch_bounds__(256) void k_flash(const _Float16* __restrict__ Q,
                                               const _Float16* __restrict__ Km,
                                               const _Float16* __restrict__ Vm,
                                               const float* __restrict__ mask,
                                               const int* __restrict__ causalp,
                                               _Float16* __restrict__ Oa) {
  constexpr int KLD = 72, VLD = 72, PLD = 72, TT = 64, NITER = Tsz / TT;
  __shared__ _Float16 sK[2][TT * KLD];   // [t][d], double-buffered
  __shared__ _Float16 sVt[2][DH * VLD];  // [d][t], double-buffered
  __shared__ _Float16 sP[8 * 16 * PLD];  // per-wave P slab [m][t]
  __shared__ float sM[2][TT];

  const int tid = threadIdx.x, lane = tid & 31, wid = tid >> 5;
  const int st = blockIdx.x, hh = blockIdx.y, bb = blockIdx.z;
  const int caus = *causalp;
  const int s0 = st * 128;
  const int hf = (lane & 16) ? 8 : 0, nl = lane & 15;

  const unsigned rowQ = (unsigned)(bb * Ssz + s0 + wid * 16);
  const _Float16* qp = Q + rowQ * CH + hh * DH;
  v16h aQ0 = frag_a(qp, CH, lane);
  v16h aQ1 = frag_a(qp + 32, CH, lane);

  v8f oacc[4];
#pragma unroll
  for (int j = 0; j < 4; ++j)
#pragma unroll
    for (int r = 0; r < 8; ++r) oacc[j][r] = 0.f;
  float mst[8], lst[8];
#pragma unroll
  for (int r = 0; r < 8; ++r) { mst[r] = -__builtin_inff(); lst[r] = 0.f; }

  const int lr = tid >> 2, lc = (tid & 3) * 16;
  const _Float16* kbase = Km + (unsigned)(bb * Tsz) * CH + hh * DH;
  const _Float16* vbase = Vm + (unsigned)(bb * Tsz) * CH + hh * DH;
  const unsigned mbase = (unsigned)(bb * Ssz) * Tsz;  // mask[b,0,*]
  _Float16* pw = sP + wid * 16 * PLD;
#if USE_TDM
  const unsigned sK_lds[2] = {(unsigned)(unsigned long long)(const void*)&sK[0][0],
                              (unsigned)(unsigned long long)(const void*)&sK[1][0]};
#endif

  // ---- prologue: tile 0 in flight ----
  v8h va0 = *(const v8h*)(vbase + (unsigned)lr * CH + lc);
  v8h va1 = *(const v8h*)(vbase + (unsigned)lr * CH + lc + 8);
  float mreg = mask[mbase + (unsigned)(tid & (TT - 1))];
#if USE_TDM
  if (tid < 32) tdm_load_2d_f16(kbase, sK_lds[0], DH, TT, 4, 3);
#else
  v8h ka0 = *(const v8h*)(kbase + (unsigned)lr * CH + lc);
  v8h ka1 = *(const v8h*)(kbase + (unsigned)lr * CH + lc + 8);
#endif

  for (int it = 0; it < NITER; ++it) {
    const int cur = it & 1;
    const int t0 = it * TT;
    _Float16* sKc = &sK[cur][0];
    _Float16* sVc = &sVt[cur][0];
    float* sMc = &sM[cur][0];

    // ---- store phase: registers -> LDS (buffer `cur` is free: barrier of
    // iteration it-1 guaranteed all waves finished compute(it-2)) ----
#pragma unroll
    for (int q = 0; q < 8; ++q) {
      sVc[(lc + q) * VLD + lr]     = va0[q];
      sVc[(lc + 8 + q) * VLD + lr] = va1[q];
    }
    if (tid < TT) sMc[tid] = mreg;
#if !USE_TDM
    *(v8h*)(sKc + lr * KLD + lc)     = ka0;
    *(v8h*)(sKc + lr * KLD + lc + 8) = ka1;
#endif
#if USE_TDM
    if (tid < 32) __builtin_amdgcn_s_wait_tensorcnt(0);  // TDM(it) done
#endif
    __syncthreads();

    // ---- issue phase: start tile it+1 (overlaps with compute below) ----
    if (it + 1 < NITER) {
      const int t1 = t0 + TT;
      const _Float16* vr = vbase + (unsigned)(t1 + lr) * CH + lc;
      va0 = *(const v8h*)vr;
      va1 = *(const v8h*)(vr + 8);
      mreg = mask[mbase + (unsigned)t1 + (unsigned)(tid & (TT - 1))];
#if USE_TDM
      if (tid < 32) {
        tdm_load_2d_f16(kbase + (unsigned)t1 * CH, sK_lds[cur ^ 1], DH, TT, 4, 3);
        if (it + 2 < NITER)
          __builtin_prefetch(kbase + (unsigned)(t1 + TT) * CH, 0, 0);  // global_prefetch_b8
      }
#else
      const _Float16* kr = kbase + (unsigned)(t1 + lr) * CH + lc;
      ka0 = *(const v8h*)kr;
      ka1 = *(const v8h*)(kr + 8);
#endif
    }

    // ---- compute phase on buffer `cur` ----
    // S = Q K^T  (16 x 64 per wave)
    v8f sacc[4];
#pragma unroll
    for (int nt = 0; nt < 4; ++nt) {
#pragma unroll
      for (int r = 0; r < 8; ++r) sacc[nt][r] = 0.f;
      v16h b0 = frag_b(sKc + nt * 16 * KLD, KLD, lane);
      v16h b1 = frag_b(sKc + nt * 16 * KLD + 32, KLD, lane);
      sacc[nt] = WMMA_F16(aQ0, b0, sacc[nt]);
      sacc[nt] = WMMA_F16(aQ1, b1, sacc[nt]);
    }

    // scale + mask
    float mc[4];
#pragma unroll
    for (int nt = 0; nt < 4; ++nt) mc[nt] = sMc[nt * 16 + nl];
#pragma unroll
    for (int nt = 0; nt < 4; ++nt) {
      const int tg = t0 + nt * 16 + nl;
#pragma unroll
      for (int r = 0; r < 8; ++r) {
        float v = sacc[nt][r] * 0.125f;  // D^-0.5
        if (mc[nt] == 0.f) v = -__builtin_inff();
        if (caus && tg > (s0 + wid * 16 + r + hf)) v = -__builtin_inff();
        sacc[nt][r] = v;
      }
    }

    // online softmax; each row lives on a 16-lane half-wave -> xor 8,4,2,1
#pragma unroll
    for (int r = 0; r < 8; ++r) {
      float mx = -__builtin_inff();
#pragma unroll
      for (int nt = 0; nt < 4; ++nt) mx = fmaxf(mx, sacc[nt][r]);
#pragma unroll
      for (int off = 8; off > 0; off >>= 1) mx = fmaxf(mx, __shfl_xor(mx, off, 32));
      const float nm = fmaxf(mst[r], mx);
      float corr = 1.f, rs = 0.f;
      if (nm == -__builtin_inff()) {
#pragma unroll
        for (int nt = 0; nt < 4; ++nt) sacc[nt][r] = 0.f;
      } else {
        corr = __expf(mst[r] - nm);
#pragma unroll
        for (int nt = 0; nt < 4; ++nt) {
          const float p = __expf(sacc[nt][r] - nm);
          sacc[nt][r] = p;
          rs += p;
        }
      }
#pragma unroll
      for (int off = 8; off > 0; off >>= 1) rs += __shfl_xor(rs, off, 32);
      lst[r] = lst[r] * corr + rs;
      mst[r] = nm;
#pragma unroll
      for (int jd = 0; jd < 4; ++jd) oacc[jd][r] *= corr;
#pragma unroll
      for (int nt = 0; nt < 4; ++nt)
        pw[(r + hf) * PLD + nt * 16 + nl] = (_Float16)sacc[nt][r];  // C->A relayout
    }

    // O += P V  (same-wave LDS ops are in order; no barrier needed for pw)
    v16h aP0 = frag_a(pw, PLD, lane);
    v16h aP1 = frag_a(pw + 32, PLD, lane);
#pragma unroll
    for (int jd = 0; jd < 4; ++jd) {
      v16h b0 = frag_b(sVc + jd * 16 * VLD, VLD, lane);
      v16h b1 = frag_b(sVc + jd * 16 * VLD + 32, VLD, lane);
      oacc[jd] = WMMA_F16(aP0, b0, oacc[jd]);
      oacc[jd] = WMMA_F16(aP1, b1, oacc[jd]);
    }
  }

  float inv[8];
#pragma unroll
  for (int r = 0; r < 8; ++r) inv[r] = (lst[r] > 0.f) ? (1.f / lst[r]) : 0.f;
  _Float16* op = Oa + rowQ * CH + hh * DH;
#pragma unroll
  for (int jd = 0; jd < 4; ++jd)
#pragma unroll
    for (int r = 0; r < 8; ++r)
      op[(unsigned)((r + hf) * CH + jd * 16 + nl)] = (_Float16)(oacc[jd][r] * inv[r]);
}

// ---------------- Output GEMM: f16 A @ f32 Wo + bo, * mask[b,s,0] -> f32 ----------------
__global__ __launch_bounds__(256) void k_gemm_out(const _Float16* __restrict__ A,
                                                  const float* __restrict__ W,
                                                  const float* __restrict__ bias,
                                                  const float* __restrict__ mask,
                                                  float* __restrict__ Out) {
  constexpr int BK = 32, ALD = 40, BLD = 40;
  __shared__ _Float16 sA[128 * ALD];
  __shared__ _Float16 sB[128 * BLD];
  const int tid = threadIdx.x, lane = tid & 31, wid = tid >> 5;
  const int wr = wid >> 1, wc = wid & 1;
  const int m0 = blockIdx.y * 128, n0 = blockIdx.x * 128;

  v8f acc[2][4];
#pragma unroll
  for (int i = 0; i < 2; ++i)
#pragma unroll
    for (int j = 0; j < 4; ++j)
#pragma unroll
      for (int r = 0; r < 8; ++r) acc[i][j][r] = 0.f;

  const int bkr = tid >> 5, bnc = (tid & 31) * 4;
#if USE_TDM
  const unsigned sA_lds = (unsigned)(unsigned long long)(const void*)&sA[0];
#else
  const int ar = tid >> 2, ac = (tid & 3) * 8;
#endif

  for (int k0 = 0; k0 < CH; k0 += BK) {
    __syncthreads();
#if USE_TDM
    if (tid < 32) {
      // 32x128 f16 tile, LDS rows 64B data + 16B pad: interval 16 DW (code 3),
      // amount 4 DW (code 3).
      tdm_load_2d_f16(A + (unsigned)(m0 * CH + k0), sA_lds, BK, 128, 3, 3);
    }
#else
#pragma unroll
    for (int rr = 0; rr < 2; ++rr) {
      const int r = ar + rr * 64;
      *(v8h*)(sA + r * ALD + ac) = *(const v8h*)(A + (unsigned)((m0 + r) * CH + k0 + ac));
    }
#endif
#pragma unroll
    for (int pp = 0; pp < 4; ++pp) {
      const int kk = bkr + pp * 8;
      const v4f w = *(const v4f*)(W + (unsigned)((k0 + kk) * CH + n0 + bnc));
      sB[(bnc + 0) * BLD + kk] = (_Float16)w[0];
      sB[(bnc + 1) * BLD + kk] = (_Float16)w[1];
      sB[(bnc + 2) * BLD + kk] = (_Float16)w[2];
      sB[(bnc + 3) * BLD + kk] = (_Float16)w[3];
    }
#if USE_TDM
    if (tid < 32) __builtin_amdgcn_s_wait_tensorcnt(0);
#endif
    __syncthreads();

    v16h aF[2], bF[4];
#pragma unroll
    for (int mt = 0; mt < 2; ++mt) aF[mt] = frag_a(sA + (wr * 32 + mt * 16) * ALD, ALD, lane);
#pragma unroll
    for (int nt = 0; nt < 4; ++nt) bF[nt] = frag_b(sB + (wc * 64 + nt * 16) * BLD, BLD, lane);
#pragma unroll
    for (int mt = 0; mt < 2; ++mt)
#pragma unroll
      for (int nt = 0; nt < 4; ++nt) acc[mt][nt] = WMMA_F16(aF[mt], bF[nt], acc[mt][nt]);
  }

  const int hf = (lane & 16) ? 8 : 0, nl = lane & 15;
#pragma unroll
  for (int mt = 0; mt < 2; ++mt)
#pragma unroll
    for (int r = 0; r < 8; ++r) {
      const int row = m0 + wr * 32 + mt * 16 + r + hf;
      const float mrow = mask[(unsigned)row * (unsigned)Tsz];  // mask[b,s,0]
#pragma unroll
      for (int nt = 0; nt < 4; ++nt) {
        const int col = n0 + wc * 64 + nt * 16 + nl;
        Out[(unsigned)(row * CH + col)] = (acc[mt][nt][r] + bias[col]) * mrow;
      }
    }
}

extern "C" void kernel_launch(void* const* d_in, const int* in_sizes, int n_in,
                              void* d_out, int out_size, void* d_ws, size_t ws_size,
                              hipStream_t stream) {
  (void)in_sizes; (void)n_in; (void)out_size; (void)ws_size;
  const float* query = (const float*)d_in[0];
  const float* key   = (const float*)d_in[1];
  const float* value = (const float*)d_in[2];
  const float* mask  = (const float*)d_in[3];
  const float* Wq = (const float*)d_in[4];
  const float* bq = (const float*)d_in[5];
  const float* Wk = (const float*)d_in[6];
  const float* bk = (const float*)d_in[7];
  const float* Wv = (const float*)d_in[8];
  const float* bv = (const float*)d_in[9];
  const float* Wo = (const float*)d_in[10];
  const float* bo = (const float*)d_in[11];
  const int* causal = (const int*)d_in[12];

  _Float16* qws = (_Float16*)d_ws;
  _Float16* kws = qws + (size_t)ROWS * CH;
  _Float16* vws = kws + (size_t)ROWS * CH;
  _Float16* aws = vws + (size_t)ROWS * CH;

  const dim3 gg(CH / 128, ROWS / 128);
  k_gemm_qkv<<<gg, 256, 0, stream>>>(query, Wq, bq, qws);
  k_gemm_qkv<<<gg, 256, 0, stream>>>(key,   Wk, bk, kws);
  k_gemm_qkv<<<gg, 256, 0, stream>>>(value, Wv, bv, vws);
  k_flash<<<dim3(Ssz / 128, NH, Bsz), 256, 0, stream>>>(qws, kws, vws, mask, causal, aws);
  k_gemm_out<<<gg, 256, 0, stream>>>(aws, Wo, bo, mask, (float*)d_out);
}
